// VectorQuantizer_34153579937809
// MI455X (gfx1250) — matrix operand order repaired and checked by reading, not compiled, not run on hardware
//
#include <hip/hip_runtime.h>
#include <hip/hip_bf16.h>

// ---------------------------------------------------------------------------
// VQ-VAE vector quantizer for MI455X (gfx1250, wave32, WMMA).
//   z_e:        (32, 256, 32, 32) f32   -> flattened rows n = b*1024 + gh*32+gw
//   emb_weight: (1024, 256) f32
// Outputs (concatenated in d_out as float):
//   [0, 8388608)            z_q_st (B,D,Gh,Gw)
//   [8388608, 8421376)      idx (as float)
//   [8421376]               vq_loss = 1.25 * mean((z_q - z)^2)
// ---------------------------------------------------------------------------

typedef __attribute__((ext_vector_type(16))) _Float16 v16h;
typedef __attribute__((ext_vector_type(8)))  float    v8f;

#define N_ROWS   32768
#define DDIM     256
#define KCODES   1024
#define PIXES    1024          // Gh*Gw
#define DPITCH   264           // padded LDS row pitch in halves (132 dwords = 4 mod 64)

// workspace layout (bytes)
#define WS_EMBH  0             // 1024*256 f16   = 524288
#define WS_E2    524288        // 1024 f32       = 4096
#define WS_IDX   528384        // 32768 i32      = 131072
#define WS_PART  659456        // 1024 f32 wave partial sums = 4096

// d_out layout (floats)
#define OUT_IDX  8388608
#define OUT_LOSS 8421376

// ---------------------------------------------------------------------------
// Kernel 1: codebook f32 -> f16, plus e2[k] = ||e_k||^2.  grid(1024) x 256
// ---------------------------------------------------------------------------
__global__ __launch_bounds__(256) void vq_prep(const float* __restrict__ emb,
                                               _Float16* __restrict__ embh,
                                               float* __restrict__ e2) {
  const int k = blockIdx.x;
  const int d = threadIdx.x;
  float w = emb[k * DDIM + d];
  embh[k * DDIM + d] = (_Float16)w;
  __shared__ float red[256];
  red[d] = w * w;
  __syncthreads();
  for (int s = 128; s > 0; s >>= 1) {
    if (d < s) red[d] += red[d + s];
    __syncthreads();
  }
  if (d == 0) e2[k] = red[0];
}

// ---------------------------------------------------------------------------
// Kernel 2: WMMA distance GEMM + argmin, double-buffered LDS code tiles.
// Block = 256 threads = 8 waves; each wave owns 16 z-rows; block owns 128 rows.
// grid(256).
// Per k-block (16 codes):
//   A (codes, 16x32 f16 per D-chunk)  <- LDS tile buf[kb&1]
//   B (z rows, 32x16 f16 per D-chunk) <- VGPRs (loaded/converted once)
//   C += A x B over 8 D-chunks; score = e2[code] - 2*dot; argmin.
// Tile kb+1 is fetched from L2 into registers while tile kb computes.
// ---------------------------------------------------------------------------
__global__ __launch_bounds__(256) void vq_dist(const float* __restrict__ z,
                                               const _Float16* __restrict__ embh,
                                               const float* __restrict__ e2,
                                               int* __restrict__ idxOut,
                                               float* __restrict__ idxF) {
  __shared__ __align__(16) _Float16 ctile[2][16 * DPITCH];
  __shared__ __align__(16) float    e2s[KCODES];

  const int tid  = threadIdx.x;
  const int wave = tid >> 5;
  const int lane = tid & 31;
  const int hi   = lane >> 4;        // which 16-lane half
  const int l16  = lane & 15;

  const int row  = blockIdx.x * 128 + wave * 16 + l16;  // this lane's z-row (N col)
  const int bb   = row >> 10;
  const int pix  = row & (PIXES - 1);
  const size_t zbase = (size_t)bb * (DDIM * PIXES) + pix;

  // --- stage e2 into LDS (first loop barrier makes it visible) -------------
  {
    const float4* s = (const float4*)e2;
    ((float4*)e2s)[tid] = s[tid];                       // 256 x 16B = 4KB
  }

  // --- B fragments: 16 z-rows x 256 D, f32 -> f16, held in VGPRs -----------
  // B layout (32x16, f16): lane<16 holds K=d 0..15 of chunk; lane>=16 K=16..31.
  v16h bfrag[8];
#pragma unroll
  for (int c = 0; c < 8; ++c) {
    const int dbase = c * 32 + hi * 16;
#pragma unroll
    for (int j = 0; j < 16; ++j) {
      bfrag[c][j] =
          (_Float16)__builtin_nontemporal_load(&z[zbase + (size_t)(dbase + j) * PIXES]);
    }
  }

  // --- staging geometry: 256 threads x 32B = one 8KB tile ------------------
  const int sr   = tid >> 4;                 // code row 0..15
  const int sseg = tid & 15;                 // 32B segment within row
  const uint4* gsrc =
      (const uint4*)(embh + (size_t)sr * DDIM + sseg * 16);  // +512 uint4 per tile
  uint4 ra0 = gsrc[0];
  uint4 ra1 = gsrc[1];
  uint4 rb0 = ra0, rb1 = ra1;

  float minS = 3.4e38f;
  int   minI = 0;

  for (int kb = 0; kb < 64; ++kb) {
    // store tile kb (fetched last iteration) into its buffer
    {
      uint4* dst = (uint4*)(&ctile[kb & 1][sr * DPITCH + sseg * 16]);
      dst[0] = ra0;
      dst[1] = ra1;
    }
    // start fetching tile kb+1 from L2 while we compute tile kb
    if (kb < 63) {
      const uint4* s2 = gsrc + (size_t)(kb + 1) * 512;
      rb0 = s2[0];
      rb1 = s2[1];
    }
    __syncthreads();   // tile kb visible; prior reads of this buffer retired

    v8f acc = {0.f, 0.f, 0.f, 0.f, 0.f, 0.f, 0.f, 0.f};
    const _Float16* tile = ctile[kb & 1];
#pragma unroll
    for (int c = 0; c < 8; ++c) {
      // A layout (16x32 f16): lane<16: M=l16, K=0..15; lane>=16 K=16..31
      const _Float16* ap = tile + l16 * DPITCH + c * 32 + hi * 16;
      v16h afrag;
      ((uint4*)&afrag)[0] = *(const uint4*)(ap);        // K +0..7
      ((uint4*)&afrag)[1] = *(const uint4*)(ap + 8);    // K +8..15
      acc = __builtin_amdgcn_wmma_f32_16x16x32_f16(
          /*neg_a=*/false, afrag, /*neg_b=*/false, bfrag[c],
          /*c_mod=*/(short)0, acc, /*reuse_a=*/false, /*reuse_b=*/false);
    }

    // C layout: VGPR v -> code = kb*16 + v + hi*8, row fixed per lane
    const int cbase = kb * 16 + hi * 8;
#pragma unroll
    for (int v = 0; v < 8; ++v) {
      const float s = e2s[cbase + v] - 2.0f * acc[v];
      if (s < minS) { minS = s; minI = cbase + v; }
    }

    ra0 = rb0;
    ra1 = rb1;
  }

  // combine the two halves (lane L and L+16 hold the same row's codes)
  const float oS = __shfl_xor(minS, 16, 32);
  const int   oI = __shfl_xor(minI, 16, 32);
  if (oS < minS || (oS == minS && oI < minI)) { minS = oS; minI = oI; }

  if (hi == 0) {
    idxOut[row] = minI;
    idxF[row]   = (float)minI;
  }
}

// ---------------------------------------------------------------------------
// Kernel 3: gather z_q (fp32, exact), write transposed output, per-wave
// partial SSE.  Wave handles 32 consecutive rows (one gw line) -> coalesced.
// grid(128) x 256  (1024 waves total).  Streams are non-temporal: z and z_q
// are touched exactly once here; keep the codebook hot instead.
// ---------------------------------------------------------------------------
__global__ __launch_bounds__(256) void vq_gather(const float* __restrict__ z,
                                                 const float* __restrict__ emb,
                                                 const int* __restrict__ idx,
                                                 float* __restrict__ out,
                                                 float* __restrict__ partial) {
  const int wid  = (blockIdx.x * 256 + threadIdx.x) >> 5;  // global wave 0..1023
  const int lane = threadIdx.x & 31;
  const int row  = wid * 32 + lane;
  const int bb   = row >> 10;
  const int pix  = row & (PIXES - 1);
  const size_t base = (size_t)bb * (DDIM * PIXES) + pix;

  const int k = idx[row];
  const float* e = emb + (size_t)k * DDIM;

  float s = 0.f;
#pragma unroll 4
  for (int d = 0; d < DDIM; ++d) {
    const float zv = __builtin_nontemporal_load(&z[base + (size_t)d * PIXES]);
    const float qv = e[d];
    __builtin_nontemporal_store(qv, &out[base + (size_t)d * PIXES]);  // ST fwd == z_q
    const float df = qv - zv;
    s += df * df;
  }
  for (int off = 16; off > 0; off >>= 1) s += __shfl_down(s, off, 32);
  if (lane == 0) partial[wid] = s;
}

// ---------------------------------------------------------------------------
// Kernel 4: deterministic serial finalize of the loss.
// ---------------------------------------------------------------------------
__global__ void vq_finalize(const float* __restrict__ partial, float* __restrict__ out) {
  if (threadIdx.x == 0 && blockIdx.x == 0) {
    float s = 0.f;
    for (int i = 0; i < 1024; ++i) s += partial[i];
    // vq_loss = mean((zq-z)^2) + BETA*mean((z-zq)^2) = 1.25 * SSE / (N*D)
    out[OUT_LOSS] = 1.25f * s / 8388608.0f;
  }
}

// ---------------------------------------------------------------------------
extern "C" void kernel_launch(void* const* d_in, const int* in_sizes, int n_in,
                              void* d_out, int out_size, void* d_ws, size_t ws_size,
                              hipStream_t stream) {
  const float* z   = (const float*)d_in[0];   // (32,256,32,32)
  const float* emb = (const float*)d_in[1];   // (1024,256)
  float* out = (float*)d_out;

  char* ws = (char*)d_ws;
  _Float16* embh   = (_Float16*)(ws + WS_EMBH);
  float*    e2     = (float*)   (ws + WS_E2);
  int*      idxWs  = (int*)     (ws + WS_IDX);
  float*    part   = (float*)   (ws + WS_PART);

  vq_prep<<<KCODES, 256, 0, stream>>>(emb, embh, e2);
  vq_dist<<<N_ROWS / 128, 256, 0, stream>>>(z, embh, e2, idxWs, out + OUT_IDX);
  vq_gather<<<N_ROWS / 256, 256, 0, stream>>>(z, emb, idxWs, out, part);
  vq_finalize<<<1, 32, 0, stream>>>(part, out);
}
